// MultiHeadAttention_55654186221755
// MI455X (gfx1250) — matrix operand order repaired and checked by reading, not compiled, not run on hardware
//
#include <hip/hip_runtime.h>

#define BB 4
#define TT 2048
#define CC 1024
#define HH 16
#define DD 64
#define MT (BB*TT)   // 8192 rows

typedef __attribute__((ext_vector_type(8)))  _Float16 v8h;
typedef __attribute__((ext_vector_type(16))) _Float16 v16h;
typedef __attribute__((ext_vector_type(8)))  float    v8f;

union V16U { v16h v; v8h h[2]; };

__device__ __forceinline__ v8f wmma_f16(v16h a, v16h b, v8f c) {
  // D = A(16x32 f16) * B(32x16 f16) + C(16x16 f32)
  return __builtin_amdgcn_wmma_f32_16x16x32_f16(
      /*neg_a=*/false, a, /*neg_b=*/false, b,
      /*c_mod=*/(short)0, c, /*reuse_a=*/false, /*reuse_b=*/false);
}

// ---------------------------------------------------------------- staging ---
__global__ void cvt_f32_to_f16(const float* __restrict__ in,
                               _Float16* __restrict__ out, int n) {
  int i = blockIdx.x * blockDim.x + threadIdx.x;
  if (i < n) out[i] = (_Float16)in[i];
}

// W[k][n] (KxN row-major, K=N=CC) -> Wt[n][k] f16
__global__ void transpose_f32_to_f16(const float* __restrict__ W,
                                     _Float16* __restrict__ Wt, int n) {
  int i = blockIdx.x * blockDim.x + threadIdx.x;
  if (i < n) {
    int row = i / CC;      // n (output row)
    int kk  = i % CC;      // k
    Wt[i] = (_Float16)W[kk * CC + row];
  }
}

// ------------------------------------------------------------------- GEMM ---
// Y = A[M,K] * Bt[N,K]^T ; each wave computes a 32(M) x 64(N) macro-tile:
// two 16-row sub-tiles reuse the same 4 B-fragments -> 8 WMMA per 12 b128
// loads per 32-wide k-step (1.5 loads/WMMA vs 2.5 for a 16x64 tile).
// mode 0: store f16 Q  [B,H,T,D] (scale folded, = 1/sqrt(D))
// mode 1: store f16 K  [B,H,T,D]
// mode 2: store f16 Vt [B,H,D,T]  (transposed for PV WMMA B-fragments)
// mode 3: store f32 out[M,N] + bias[n]
__global__ void __launch_bounds__(256) gemm_wmma_32x64(
    const _Float16* __restrict__ A, const _Float16* __restrict__ Bt,
    int M, int N, int K, int mode, float scale,
    _Float16* __restrict__ outh, float* __restrict__ outf,
    const float* __restrict__ bias)
{
  int lane = threadIdx.x & 31;
  int wave = blockIdx.x * (blockDim.x >> 5) + (threadIdx.x >> 5);
  int tilesN = N / 64;
  int tm = wave / tilesN, tn = wave % tilesN;
  if (tm >= M / 32) return;
  int m0 = tm * 32, n0 = tn * 64;
  int hi = lane >> 4, col = lane & 15;

  v8f acc[2][4] = {};
  const _Float16* arow0 = A + (size_t)(m0 + col) * K;       // sub-tile 0 row
  const _Float16* arow1 = A + (size_t)(m0 + 16 + col) * K;  // sub-tile 1 row
  for (int kc = 0; kc < K; kc += 32) {
    V16U a0, a1;  // A-fragment layout: K = hi*16..+15
    a0.h[0] = *(const v8h*)(arow0 + kc + hi * 16);
    a0.h[1] = *(const v8h*)(arow0 + kc + hi * 16 + 8);
    a1.h[0] = *(const v8h*)(arow1 + kc + hi * 16);
    a1.h[1] = *(const v8h*)(arow1 + kc + hi * 16 + 8);
#pragma unroll
    for (int g = 0; g < 4; ++g) {
      const _Float16* brow = Bt + (size_t)(n0 + g * 16 + col) * K;
      v16h bf = *(const v16h*)(brow + kc + hi * 16);  // B: K = hi*16..+15
      acc[0][g] = wmma_f16(a0.v, bf, acc[0][g]);
      acc[1][g] = wmma_f16(a1.v, bf, acc[1][g]);
    }
  }
#pragma unroll
  for (int s = 0; s < 2; ++s) {
#pragma unroll
    for (int g = 0; g < 4; ++g) {
#pragma unroll
      for (int r = 0; r < 8; ++r) {
        int m  = m0 + s * 16 + r + hi * 8;  // C/D: row = r + 8*hi
        int nn = n0 + g * 16 + col;         //      col = lane&15
        float val = acc[s][g][r] * scale;
        if (mode == 3) {
          outf[(size_t)m * N + nn] = val + bias[nn];
        } else {
          int bb = m / TT, t = m % TT;
          int hh = nn / DD, d = nn % DD;
          if (mode == 2)
            outh[(((size_t)bb * HH + hh) * DD + d) * TT + t] = (_Float16)val;
          else
            outh[(((size_t)bb * HH + hh) * TT + t) * DD + d] = (_Float16)val;
        }
      }
    }
  }
}

// -------------------------------------------------------- flash attention ---
// One wave per (b, h, 16-query tile). Iterates 32-key tiles: 4 WMMA for
// S = Q*K^T, online softmax (shfl reductions within 16-lane halves), P is
// bounced via a private LDS tile to convert D-layout -> A-layout, then
// 4 WMMA accumulate O += P*V. Vt is [D,T] so B-fragments are contiguous.
__global__ void __launch_bounds__(128) flash_attn_wmma(
    const _Float16* __restrict__ Qh,   // [B,H,T,D], pre-scaled by 1/sqrt(D)
    const _Float16* __restrict__ Kh,   // [B,H,T,D]
    const _Float16* __restrict__ Vt,   // [B,H,D,T]
    _Float16* __restrict__ attn)       // [B,T,C] f16
{
  __shared__ alignas(32) _Float16 Pl[4][16 * 32];
  int lane  = threadIdx.x & 31;
  int wslot = threadIdx.x >> 5;
  int wave  = blockIdx.x * 4 + wslot;
  int qt = wave % (TT / 16);
  int bh = wave / (TT / 16);
  int q0 = qt * 16;
  const _Float16* Qhead = Qh + (size_t)bh * TT * DD;
  const _Float16* Khead = Kh + (size_t)bh * TT * DD;
  const _Float16* Vhead = Vt + (size_t)bh * DD * TT;
  int hi = lane >> 4, col = lane & 15;

  // Q A-fragments for D=64 split into two K=32 chunks
  const _Float16* qrow = Qhead + (size_t)(q0 + col) * DD;
  V16U a0, a1;
  a0.h[0] = *(const v8h*)(qrow + hi * 16);
  a0.h[1] = *(const v8h*)(qrow + 8 + hi * 16);
  a1.h[0] = *(const v8h*)(qrow + 32 + hi * 16);
  a1.h[1] = *(const v8h*)(qrow + 40 + hi * 16);

  v8f O[4] = {};
  float rowmax[8], rowsum[8];
#pragma unroll
  for (int r = 0; r < 8; ++r) { rowmax[r] = -3.0e38f; rowsum[r] = 0.0f; }
  _Float16* P = &Pl[wslot][0];

  int ktiles = (q0 + 16 + 31) / 32;   // causal: only tiles with k0 <= q0+15
  for (int kt = 0; kt < ktiles; ++kt) {
    int k0 = kt * 32;
    v8f S0 = {}, S1 = {};
#pragma unroll
    for (int c = 0; c < 2; ++c) {     // contraction over D in chunks of 32
      v16h b0 = *(const v16h*)(Khead + (size_t)(k0 + col) * DD + c * 32 + hi * 16);
      v16h b1 = *(const v16h*)(Khead + (size_t)(k0 + 16 + col) * DD + c * 32 + hi * 16);
      v16h a = c ? a1.v : a0.v;
      S0 = wmma_f16(a, b0, S0);
      S1 = wmma_f16(a, b1, S1);
    }
#pragma unroll
    for (int r = 0; r < 8; ++r) {
      int rm  = r + hi * 8;
      int row = q0 + rm;
      float s0 = (k0 + col      <= row) ? S0[r] : -3.0e38f;  // causal mask
      float s1 = (k0 + 16 + col <= row) ? S1[r] : -3.0e38f;
      float mx = fmaxf(s0, s1);
      mx = fmaxf(mx, __shfl_xor(mx, 1, 32));
      mx = fmaxf(mx, __shfl_xor(mx, 2, 32));
      mx = fmaxf(mx, __shfl_xor(mx, 4, 32));
      mx = fmaxf(mx, __shfl_xor(mx, 8, 32));
      float newm = fmaxf(rowmax[r], mx);
      float fac  = __expf(rowmax[r] - newm);
      float p0 = __expf(s0 - newm);
      float p1 = __expf(s1 - newm);
      float ps = p0 + p1;
      ps += __shfl_xor(ps, 1, 32);
      ps += __shfl_xor(ps, 2, 32);
      ps += __shfl_xor(ps, 4, 32);
      ps += __shfl_xor(ps, 8, 32);
      rowsum[r] = rowsum[r] * fac + ps;
      rowmax[r] = newm;
#pragma unroll
      for (int g = 0; g < 4; ++g) O[g][r] *= fac;
      P[rm * 32 + col]      = (_Float16)p0;   // 16x32 f16 tile, row-major
      P[rm * 32 + 16 + col] = (_Float16)p1;
    }
    // LDS stores -> loads within this wave only; wait for stores to land.
    asm volatile("s_wait_dscnt 0" ::: "memory");
    V16U ap;  // A-fragment of P: row = lane&15, K = hi*16..+15
    ap.h[0] = *(const v8h*)(P + col * 32 + hi * 16);
    ap.h[1] = *(const v8h*)(P + col * 32 + hi * 16 + 8);
#pragma unroll
    for (int g = 0; g < 4; ++g) {
      v16h bv = *(const v16h*)(Vhead + (size_t)(g * 16 + col) * TT + k0 + hi * 16);
      O[g] = wmma_f16(ap.v, bv, O[g]);
    }
  }

  int b = bh / HH, h = bh % HH;
#pragma unroll
  for (int r = 0; r < 8; ++r) {
    int rm = r + hi * 8;
    float inv = 1.0f / rowsum[r];
    size_t base = ((size_t)b * TT + q0 + rm) * CC + h * DD;
#pragma unroll
    for (int g = 0; g < 4; ++g)
      attn[base + g * 16 + col] = (_Float16)(O[g][r] * inv);
  }
}

// ----------------------------------------------------------------- launch ---
extern "C" void kernel_launch(void* const* d_in, const int* in_sizes, int n_in,
                              void* d_out, int out_size, void* d_ws, size_t ws_size,
                              hipStream_t stream) {
  (void)in_sizes; (void)n_in; (void)out_size; (void)ws_size;
  const float* x  = (const float*)d_in[0];
  const float* Wq = (const float*)d_in[1];
  const float* Wk = (const float*)d_in[2];
  const float* Wv = (const float*)d_in[3];
  const float* Wo = (const float*)d_in[4];
  const float* bo = (const float*)d_in[5];
  float* out = (float*)d_out;

  char* ws = (char*)d_ws;
  size_t off = 0;
  auto take = [&](size_t nbytes) -> void* {
    void* p = ws + off;
    off += (nbytes + 255) & ~(size_t)255;
    return p;
  };
  _Float16* xh   = (_Float16*)take((size_t)MT * CC * 2);        // 16 MB
  _Float16* Wqt  = (_Float16*)take((size_t)CC * CC * 2);        // 2 MB
  _Float16* Wkt  = (_Float16*)take((size_t)CC * CC * 2);
  _Float16* Wvt  = (_Float16*)take((size_t)CC * CC * 2);
  _Float16* Wot  = (_Float16*)take((size_t)CC * CC * 2);
  _Float16* Qh   = (_Float16*)take((size_t)BB * HH * TT * DD * 2); // 16 MB
  _Float16* Kh   = (_Float16*)take((size_t)BB * HH * TT * DD * 2);
  _Float16* Vth  = (_Float16*)take((size_t)BB * HH * DD * TT * 2);
  _Float16* attn = (_Float16*)take((size_t)MT * CC * 2);        // 16 MB

  { // stage x -> f16
    int n = MT * CC;
    cvt_f32_to_f16<<<(n + 255) / 256, 256, 0, stream>>>(x, xh, n);
  }
  { // stage weights -> transposed f16
    int n = CC * CC;
    int g = (n + 255) / 256;
    transpose_f32_to_f16<<<g, 256, 0, stream>>>(Wq, Wqt, n);
    transpose_f32_to_f16<<<g, 256, 0, stream>>>(Wk, Wkt, n);
    transpose_f32_to_f16<<<g, 256, 0, stream>>>(Wv, Wvt, n);
    transpose_f32_to_f16<<<g, 256, 0, stream>>>(Wo, Wot, n);
  }
  { // Q/K/V projections (Q pre-scaled by 1/sqrt(D)=0.125)
    int waves  = (MT / 32) * (CC / 64);   // 4096 waves
    int blocks = waves / 8;               // 256 thr = 8 waves/block
    gemm_wmma_32x64<<<blocks, 256, 0, stream>>>(xh, Wqt, MT, CC, CC, 0, 0.125f, Qh,  nullptr, nullptr);
    gemm_wmma_32x64<<<blocks, 256, 0, stream>>>(xh, Wkt, MT, CC, CC, 1, 1.0f,   Kh,  nullptr, nullptr);
    gemm_wmma_32x64<<<blocks, 256, 0, stream>>>(xh, Wvt, MT, CC, CC, 2, 1.0f,   Vth, nullptr, nullptr);
  }
  { // causal flash attention
    int waves = BB * HH * (TT / 16);      // 8192 waves, 4 per block
    flash_attn_wmma<<<waves / 4, 128, 0, stream>>>(Qh, Kh, Vth, attn);
  }
  { // output projection + bias, f32 out
    int waves = (MT / 32) * (CC / 64);
    gemm_wmma_32x64<<<waves / 8, 256, 0, stream>>>(attn, Wot, MT, CC, CC, 3, 1.0f, nullptr, out, bo);
  }
}
